// ConvModel_24412594111221
// MI455X (gfx1250) — matrix operand order repaired and checked by reading, not compiled, not run on hardware
//
#include <hip/hip_runtime.h>
#include <hip/hip_bf16.h>
#include <stdint.h>

#define EPSF 1e-5f
#define BATCH 4096

typedef __attribute__((ext_vector_type(16))) _Float16 v16h;
typedef __attribute__((ext_vector_type(8)))  _Float16 v8h;
typedef __attribute__((ext_vector_type(8)))  float    v8f;

__device__ __forceinline__ float sgnf(float v) { return v >= 0.0f ? 1.0f : -1.0f; }

// ---------- CDNA5 async global->LDS copy (tracked by ASYNCcnt) ----------
__device__ __forceinline__ void async_g2l_b128(void* lds, const void* gsrc) {
  uint32_t l = (uint32_t)(uintptr_t)lds;   // low 32 bits of generic ptr = LDS offset
  asm volatile("global_load_async_to_lds_b128 %0, %1, off"
               :: "v"(l), "v"(gsrc) : "memory");
}
// async loads retire in order -> waiting for <=N leaves the newest N in flight
__device__ __forceinline__ void async_wait0() {
  asm volatile("s_wait_asynccnt 0" ::: "memory");
}
__device__ __forceinline__ void async_wait5() {
  asm volatile("s_wait_asynccnt 5" ::: "memory");
}
__device__ __forceinline__ void async_wait3() {
  asm volatile("s_wait_asynccnt 3" ::: "memory");
}

// ---------- WMMA helpers (16x16x32 f16 -> f32) ----------
__device__ __forceinline__ v8f wmma16x16x32(v16h a, v16h b, v8f c) {
  return __builtin_amdgcn_wmma_f32_16x16x32_f16(false, a, false, b, (short)0, c,
                                                false, false);
}

// A-matrix 16x32 f16 layout (ISA 7.12.2): lane l -> row m=l&15;
// holds K = [ko..ko+7] and [ko+16..ko+23], ko = (l>>4)*8. LDS stored row-major [m][k].
__device__ __forceinline__ v16h frag_a(const _Float16* s, int ld) {
  const int l = threadIdx.x & 31;
  const _Float16* p = s + (l & 15) * ld + ((l >> 4) << 3);
  v8h lo = *(const v8h*)(p);
  v8h hi = *(const v8h*)(p + 16);
  v16h r;
#pragma unroll
  for (int i = 0; i < 8; ++i) { r[i] = lo[i]; r[i + 8] = hi[i]; }
  return r;
}

// B-matrix 32x16 f16: lane l -> col n=l&15; K = [ko..ko+7] and [ko+16..ko+23], ko=(l>>4)*8.
// LDS stored [n][k] (K contiguous per column), which matches im2col row layout.
__device__ __forceinline__ v16h frag_b(const _Float16* s, int ld) {
  const int l = threadIdx.x & 31;
  const _Float16* p = s + (l & 15) * ld + ((l >> 4) << 3);
  v8h lo = *(const v8h*)(p);
  v8h hi = *(const v8h*)(p + 16);
  v16h r;
#pragma unroll
  for (int i = 0; i < 8; ++i) { r[i] = lo[i]; r[i + 8] = hi[i]; }
  return r;
}

// ---------- K0: binarize conv2/fc1 weights to fp16 (+-1) ----------
// w2h[m][k], k = (ky*5+kx)*64 + c  (im2col-K order, channels contiguous)
__global__ void binarize_weights_kernel(const float* __restrict__ c2w,
                                        const float* __restrict__ f1w,
                                        _Float16* __restrict__ w2h,
                                        _Float16* __restrict__ w1h) {
  const int N2 = 96 * 1600, N1 = 1024 * 1536;
  for (int i = blockIdx.x * blockDim.x + threadIdx.x; i < N2 + N1;
       i += gridDim.x * blockDim.x) {
    if (i < N2) {
      int m = i / 1600, k = i % 1600;
      int tap = k >> 6, c = k & 63;
      w2h[i] = (_Float16)sgnf(c2w[(m * 64 + c) * 25 + tap]);
    } else {
      int j = i - N2;
      w1h[j] = (_Float16)sgnf(f1w[j]);
    }
  }
}

// ---------- K1: conv1 (1->64, 5x5) + sign-pack + positive counts ----------
__global__ void __launch_bounds__(256) conv1_kernel(
    const float* __restrict__ x, const float* __restrict__ w,
    const float* __restrict__ bias, uint32_t* __restrict__ sbits,
    int* __restrict__ counts) {
  __shared__ float sw[64 * 25];
  __shared__ int cnt[64];
  const int tid = threadIdx.x;
  for (int i = tid; i < 1600; i += 256) sw[i] = sgnf(w[i]);
  if (tid < 64) cnt[tid] = 0;
  __syncthreads();
  const int p = blockIdx.x * 256 + tid;        // p < B*576 exactly
  const int b = p / 576, r = p % 576;
  const int oy = r / 24, ox = r % 24;
  float win[25];
  const float* xb = x + (size_t)b * 784;
#pragma unroll
  for (int ky = 0; ky < 5; ++ky)
#pragma unroll
    for (int kx = 0; kx < 5; ++kx)
      win[ky * 5 + kx] = xb[(oy + ky) * 28 + ox + kx];
  uint32_t bits0 = 0, bits1 = 0;
  for (int c = 0; c < 64; ++c) {
    float acc = bias[c];
    const float* wc = &sw[c * 25];
#pragma unroll
    for (int j = 0; j < 25; ++j) acc += wc[j] * win[j];
    int pos = acc >= 0.0f;
    if (c < 32) bits0 |= (uint32_t)pos << c;
    else        bits1 |= (uint32_t)pos << (c - 32);
    if (pos) atomicAdd(&cnt[c], 1);
  }
  sbits[(size_t)p * 2]     = bits0;
  sbits[(size_t)p * 2 + 1] = bits1;
  __syncthreads();
  if (tid < 64) atomicAdd(&counts[tid], cnt[tid]);
}

// ---------- BN over +-1 tensor: from positive counts ----------
__global__ void bn_sign_params_kernel(const int* __restrict__ counts,
                                      const float* __restrict__ g,
                                      const float* __restrict__ bb,
                                      float* __restrict__ sc, float* __restrict__ sh,
                                      int C, float T) {
  int c = blockIdx.x * blockDim.x + threadIdx.x;
  if (c >= C) return;
  float m = (2.0f * (float)counts[c] - T) / T;   // mean of +-1 values
  float var = 1.0f - m * m;                      // E[x^2]=1
  float r = rsqrtf(var + EPSF);
  sc[c] = g[c] * r;
  sh[c] = bb[c] - m * g[c] * r;
}

// ---------- K3: bn1 + maxpool 2x2 -> h2 (B,12,12,64) fp16 channel-last ----------
__global__ void __launch_bounds__(256) pool1_kernel(
    const uint32_t* __restrict__ sbits, const float* __restrict__ sc1,
    const float* __restrict__ sh1, _Float16* __restrict__ h2) {
  __shared__ float sc[64], sh[64];
  if (threadIdx.x < 64) { sc[threadIdx.x] = sc1[threadIdx.x]; sh[threadIdx.x] = sh1[threadIdx.x]; }
  __syncthreads();
  const int p = blockIdx.x * 256 + threadIdx.x;  // p < B*144 exactly
  const int b = p / 144, r = p % 144;
  const int y = r / 12, xx = r % 12;
  uint32_t w0[4], w1[4];
#pragma unroll
  for (int d = 0; d < 4; ++d) {
    int iy = 2 * y + (d >> 1), ix = 2 * xx + (d & 1);
    size_t q = ((size_t)b * 576 + iy * 24 + ix) * 2;
    w0[d] = sbits[q]; w1[d] = sbits[q + 1];
  }
  _Float16* o = h2 + (size_t)p * 64;
  for (int c = 0; c < 64; ++c) {
    float scc = sc[c], shc = sh[c];
    float m = -1e30f;
#pragma unroll
    for (int d = 0; d < 4; ++d) {
      uint32_t word = (c < 32) ? w0[d] : w1[d];
      float s = ((word >> (c & 31)) & 1u) ? 1.0f : -1.0f;
      m = fmaxf(m, shc + s * scc);
    }
    o[c] = (_Float16)m;
  }
}

// ---------- K4: conv2 implicit GEMM via WMMA (double-buffered async LDS) ----------
// M=96 out channels, N = 64 pixels per image (one image per block), K=1600.
// Block = 128 threads (4 waves); wave w owns N-tile w; 6 M-tiles each.
// Each thread issues 5 async b128 copies per K-stage (3 for A, 2 for B).
__device__ __forceinline__ void conv2_stage(const _Float16* __restrict__ h2,
                                            const _Float16* __restrict__ w2h,
                                            _Float16* As, _Float16* Bs,
                                            int kc, int p0, int tid) {
  const int tap = kc >> 1;
  const int c0 = (kc & 1) << 5;
  const int ky = tap / 5, kx = tap % 5;
  // stage A: 96x32 halfs = 384 chunks of 16B
#pragma unroll
  for (int s = 0; s < 3; ++s) {
    int ch = tid + s * 128;
    int row = ch >> 2, q = ch & 3;
    async_g2l_b128(&As[row * 40 + q * 8],
                   w2h + (size_t)row * 1600 + kc * 32 + q * 8);
  }
  // stage B: im2col gather, 64x32 halfs = 256 chunks of 16B
#pragma unroll
  for (int s = 0; s < 2; ++s) {
    int ch = tid + s * 128;
    int pix = p0 + (ch >> 2), q = ch & 3;
    int b = pix >> 6, oy = (pix >> 3) & 7, ox = pix & 7;
    size_t g = ((size_t)(b * 12 + oy + ky) * 12 + (ox + kx)) * 64 + c0 + q * 8;
    async_g2l_b128(&Bs[(ch >> 2) * 40 + q * 8], h2 + g);
  }
}

__global__ void __launch_bounds__(128) conv2_wmma_kernel(
    const _Float16* __restrict__ h2, const _Float16* __restrict__ w2h,
    const float* __restrict__ c2b, float* __restrict__ y2) {
  __shared__ __align__(16) _Float16 As[2][96 * 40];   // [m][k], ld=40 (pad)
  __shared__ __align__(16) _Float16 Bs[2][64 * 40];   // [pixel][k]
  const int tid = threadIdx.x;
  const int wave = tid >> 5;
  const int p0 = blockIdx.x * 64;                     // pixel = b*64 + oy*8 + ox
  v8f acc[6] = {};
  conv2_stage(h2, w2h, As[0], Bs[0], 0, p0, tid);     // prologue: stage 0 in flight
  for (int kc = 0; kc < 50; ++kc) {
    const int cur = kc & 1;
    if (kc + 1 < 50) {
      conv2_stage(h2, w2h, As[cur ^ 1], Bs[cur ^ 1], kc + 1, p0, tid);
      __builtin_prefetch(w2h + (size_t)(kc + 1) * 32, 0, 1);
      async_wait5();            // stage kc landed; stage kc+1 still streaming
    } else {
      async_wait0();
    }
    __syncthreads();            // all waves see stage kc's LDS data
    // hoist all fragments -> one dscnt wait, then back-to-back WMMAs
    v16h bf = frag_b(&Bs[cur][wave * 16 * 40], 40);
    v16h af[6];
#pragma unroll
    for (int mt = 0; mt < 6; ++mt) af[mt] = frag_a(&As[cur][mt * 16 * 40], 40);
#pragma unroll
    for (int mt = 0; mt < 6; ++mt) acc[mt] = wmma16x16x32(af[mt], bf, acc[mt]);
    __syncthreads();            // done reading buf before it is refilled
  }
  // D layout: lane l -> col n=l&15, VGPR i -> row (l>>4)*8 + i
  const int lane = tid & 31;
  const int n = lane & 15, mrow = (lane >> 4) << 3;
  const size_t pix = (size_t)p0 + wave * 16 + n;
#pragma unroll
  for (int mt = 0; mt < 6; ++mt) {
    int m0 = mt * 16 + mrow;
    float* o = y2 + pix * 96 + m0;
#pragma unroll
    for (int i = 0; i < 8; ++i) o[i] = acc[mt][i] + c2b[m0 + i];
  }
}

// ---------- generic positive-count kernel (c = i % C) ----------
__global__ void __launch_bounds__(256) count_pos_kernel(
    const float* __restrict__ src, long n, int C, int* __restrict__ counts) {
  __shared__ int cnt[1024];
  for (int i = threadIdx.x; i < C; i += 256) cnt[i] = 0;
  __syncthreads();
  long i = (long)blockIdx.x * 256 + threadIdx.x;
  const long stride = (long)gridDim.x * 256;
  for (; i < n; i += stride)
    if (src[i] >= 0.0f) atomicAdd(&cnt[(int)(i % C)], 1);
  __syncthreads();
  for (int c = threadIdx.x; c < C; c += 256)
    if (cnt[c]) atomicAdd(&counts[c], cnt[c]);
}

// ---------- pool2 value: sign(y2) -> bn2 two-value map -> max 2x2 ----------
__device__ __forceinline__ float pool2_val(const float* __restrict__ y2,
                                           int b, int f, float sc, float sh) {
  int c = f >> 4, py = (f >> 2) & 3, px = f & 3;
  float m = -1e30f;
#pragma unroll
  for (int dy = 0; dy < 2; ++dy)
#pragma unroll
    for (int dx = 0; dx < 2; ++dx) {
      float v = y2[((size_t)(b * 8 + 2 * py + dy) * 8 + (2 * px + dx)) * 96 + c];
      m = fmaxf(m, sh + sgnf(v) * sc);
    }
  return m;
}

// ---------- K7: per-feature sum / sumsq over batch (one block per feature) ----------
__global__ void __launch_bounds__(256) feat_stats_kernel(
    const float* __restrict__ y2, const float* __restrict__ sc2,
    const float* __restrict__ sh2, float* __restrict__ sum3,
    float* __restrict__ ssq3) {
  const int f = blockIdx.x;
  const int c = f >> 4;
  const float sc = sc2[c], sh = sh2[c];
  float s = 0.0f, q = 0.0f;
  for (int b = threadIdx.x; b < BATCH; b += 256) {
    float v = pool2_val(y2, b, f, sc, sh);
    s += v; q += v * v;
  }
  __shared__ float rs[256], rq[256];
  rs[threadIdx.x] = s; rq[threadIdx.x] = q;
  __syncthreads();
  for (int off = 128; off > 0; off >>= 1) {
    if (threadIdx.x < off) {
      rs[threadIdx.x] += rs[threadIdx.x + off];
      rq[threadIdx.x] += rq[threadIdx.x + off];
    }
    __syncthreads();
  }
  if (threadIdx.x == 0) { sum3[f] = rs[0]; ssq3[f] = rq[0]; }
}

__global__ void bn3_params_kernel(const float* __restrict__ sum3,
                                  const float* __restrict__ ssq3,
                                  const float* __restrict__ g,
                                  const float* __restrict__ bb,
                                  float* __restrict__ sc, float* __restrict__ sh) {
  int f = blockIdx.x * blockDim.x + threadIdx.x;
  if (f >= 1536) return;
  float mean = sum3[f] * (1.0f / BATCH);
  float var = ssq3[f] * (1.0f / BATCH) - mean * mean;
  float r = rsqrtf(var + EPSF);
  sc[f] = g[f] * r;
  sh[f] = bb[f] - mean * g[f] * r;
}

// ---------- K9: apply bn3, emit fc1 activations as fp16 (B,1536) ----------
__global__ void __launch_bounds__(256) bn3_apply_kernel(
    const float* __restrict__ y2, const float* __restrict__ sc2,
    const float* __restrict__ sh2, const float* __restrict__ sc3,
    const float* __restrict__ sh3, _Float16* __restrict__ h3) {
  int i = blockIdx.x * 256 + threadIdx.x;
  if (i >= BATCH * 1536) return;
  int b = i / 1536, f = i % 1536;
  float v = pool2_val(y2, b, f, sc2[f >> 4], sh2[f >> 4]);
  h3[i] = (_Float16)(v * sc3[f] + sh3[f]);
}

// ---------- K10: fc1 GEMM via WMMA (double-buffered async LDS) ----------
// M=1024, N=4096, K=1536. Block tile 128(M) x 64(N), 256 threads = 8 waves;
// wave -> (mg = w>>2, nt = w&3), each wave computes 4 M-tiles x 1 N-tile.
// Each thread issues 3 async b128 copies per K-stage (2 for W, 1 for H).
__device__ __forceinline__ void fc1_stage(const _Float16* __restrict__ h3,
                                          const _Float16* __restrict__ w1h,
                                          _Float16* Ws, _Float16* Hs,
                                          int kc, int mblk, int nblk, int tid) {
#pragma unroll
  for (int s = 0; s < 2; ++s) {
    int ch = tid + s * 256;
    int row = ch >> 2, q = ch & 3;
    async_g2l_b128(&Ws[row * 40 + q * 8],
                   w1h + (size_t)(mblk * 128 + row) * 1536 + kc * 32 + q * 8);
  }
  {
    int n = tid >> 2, q = tid & 3;
    async_g2l_b128(&Hs[n * 40 + q * 8],
                   h3 + (size_t)(nblk * 64 + n) * 1536 + kc * 32 + q * 8);
  }
}

__global__ void __launch_bounds__(256) fc1_wmma_kernel(
    const _Float16* __restrict__ h3, const _Float16* __restrict__ w1h,
    const float* __restrict__ b1, float* __restrict__ y3) {
  __shared__ __align__(16) _Float16 Ws[2][128 * 40];
  __shared__ __align__(16) _Float16 Hs[2][64 * 40];
  const int tid = threadIdx.x;
  const int wave = tid >> 5;
  const int nt = wave & 3, mg = wave >> 2;
  const int mblk = blockIdx.x >> 6;   // 0..7
  const int nblk = blockIdx.x & 63;   // 0..63
  v8f acc[4] = {};
  fc1_stage(h3, w1h, Ws[0], Hs[0], 0, mblk, nblk, tid);
  for (int kc = 0; kc < 48; ++kc) {
    const int cur = kc & 1;
    if (kc + 1 < 48) {
      fc1_stage(h3, w1h, Ws[cur ^ 1], Hs[cur ^ 1], kc + 1, mblk, nblk, tid);
      async_wait3();
    } else {
      async_wait0();
    }
    __syncthreads();
    v16h bf = frag_b(&Hs[cur][nt * 16 * 40], 40);
    v16h af[4];
#pragma unroll
    for (int i = 0; i < 4; ++i) af[i] = frag_a(&Ws[cur][(mg * 4 + i) * 16 * 40], 40);
#pragma unroll
    for (int i = 0; i < 4; ++i) acc[i] = wmma16x16x32(af[i], bf, acc[i]);
    __syncthreads();
  }
  const int lane = tid & 31;
  const int n = lane & 15, mrow = (lane >> 4) << 3;
  const size_t b = (size_t)nblk * 64 + nt * 16 + n;
#pragma unroll
  for (int i = 0; i < 4; ++i) {
    int o0 = mblk * 128 + (mg * 4 + i) * 16 + mrow;
    float* o = y3 + b * 1024 + o0;
#pragma unroll
    for (int j = 0; j < 8; ++j) o[j] = acc[i][j] + b1[o0 + j];
  }
}

// ---------- K12: bn4(sign(y3)) @ sign(fc2_w)^T + b ----------
__global__ void __launch_bounds__(256) fc2_kernel(
    const float* __restrict__ y3, const float* __restrict__ sc4,
    const float* __restrict__ sh4, const float* __restrict__ w,
    const float* __restrict__ bias, float* __restrict__ out) {
  __shared__ float s4[1024], h4[1024];
  for (int i = threadIdx.x; i < 1024; i += 256) { s4[i] = sc4[i]; h4[i] = sh4[i]; }
  __syncthreads();
  int idx = blockIdx.x * 256 + threadIdx.x;
  if (idx >= BATCH * 10) return;
  int b = idx / 10, m = idx % 10;
  const float* yb = y3 + (size_t)b * 1024;
  const float* wm = w + (size_t)m * 1024;
  float acc = bias[m];
  for (int k = 0; k < 1024; ++k) {
    float h = sgnf(yb[k]) * s4[k] + h4[k];
    acc += (wm[k] >= 0.0f) ? h : -h;
  }
  out[idx] = acc;
}

extern "C" void kernel_launch(void* const* d_in, const int* in_sizes, int n_in,
                              void* d_out, int out_size, void* d_ws, size_t ws_size,
                              hipStream_t stream) {
  const float* x    = (const float*)d_in[0];
  const float* c1w  = (const float*)d_in[1];
  const float* c1b  = (const float*)d_in[2];
  const float* bn1g = (const float*)d_in[3];
  const float* bn1b = (const float*)d_in[4];
  const float* c2w  = (const float*)d_in[5];
  const float* c2b  = (const float*)d_in[6];
  const float* bn2g = (const float*)d_in[7];
  const float* bn2b = (const float*)d_in[8];
  const float* bn3g = (const float*)d_in[9];
  const float* bn3b = (const float*)d_in[10];
  const float* f1w  = (const float*)d_in[11];
  const float* f1b  = (const float*)d_in[12];
  const float* bn4g = (const float*)d_in[13];
  const float* bn4b = (const float*)d_in[14];
  const float* f2w  = (const float*)d_in[15];
  const float* f2b  = (const float*)d_in[16];
  float* out = (float*)d_out;
  char* ws = (char*)d_ws;

  size_t off = 0;
  auto alloc = [&](size_t bytes) {
    size_t r = off; off += (bytes + 255) & ~(size_t)255; return r;
  };
  size_t o_cnt  = alloc(1184 * 4);                 // counts1[64]|counts2[96]|counts4[1024]
  size_t o_sc1  = alloc(64 * 4),   o_sh1 = alloc(64 * 4);
  size_t o_sc2  = alloc(96 * 4),   o_sh2 = alloc(96 * 4);
  size_t o_sc3  = alloc(1536 * 4), o_sh3 = alloc(1536 * 4);
  size_t o_sc4  = alloc(1024 * 4), o_sh4 = alloc(1024 * 4);
  size_t o_sum3 = alloc(1536 * 4), o_ssq3 = alloc(1536 * 4);
  size_t o_w2h  = alloc((size_t)96 * 1600 * 2);
  size_t o_w1h  = alloc((size_t)1024 * 1536 * 2);
  size_t o_sb1  = alloc((size_t)BATCH * 576 * 8);
  size_t o_h2   = alloc((size_t)BATCH * 144 * 64 * 2);
  size_t o_y2   = alloc((size_t)BATCH * 64 * 96 * 4);
  size_t o_h3   = alloc((size_t)BATCH * 1536 * 2);
  size_t o_y3   = alloc((size_t)BATCH * 1024 * 4);

  int* counts1 = (int*)(ws + o_cnt);
  int* counts2 = counts1 + 64;
  int* counts4 = counts1 + 160;
  float* sc1 = (float*)(ws + o_sc1); float* sh1 = (float*)(ws + o_sh1);
  float* sc2 = (float*)(ws + o_sc2); float* sh2 = (float*)(ws + o_sh2);
  float* sc3 = (float*)(ws + o_sc3); float* sh3 = (float*)(ws + o_sh3);
  float* sc4 = (float*)(ws + o_sc4); float* sh4 = (float*)(ws + o_sh4);
  float* sum3 = (float*)(ws + o_sum3); float* ssq3 = (float*)(ws + o_ssq3);
  _Float16* w2h = (_Float16*)(ws + o_w2h);
  _Float16* w1h = (_Float16*)(ws + o_w1h);
  uint32_t* sb1 = (uint32_t*)(ws + o_sb1);
  _Float16* h2  = (_Float16*)(ws + o_h2);
  float*    y2  = (float*)(ws + o_y2);
  _Float16* h3  = (_Float16*)(ws + o_h3);
  float*    y3  = (float*)(ws + o_y3);

  hipMemsetAsync(ws + o_cnt, 0, 1184 * 4, stream);

  binarize_weights_kernel<<<4096, 256, 0, stream>>>(c2w, f1w, w2h, w1h);
  conv1_kernel<<<9216, 256, 0, stream>>>(x, c1w, c1b, sb1, counts1);
  bn_sign_params_kernel<<<1, 256, 0, stream>>>(counts1, bn1g, bn1b, sc1, sh1,
                                               64, 4096.0f * 576.0f);
  pool1_kernel<<<2304, 256, 0, stream>>>(sb1, sc1, sh1, h2);
  conv2_wmma_kernel<<<4096, 128, 0, stream>>>(h2, w2h, c2b, y2);
  count_pos_kernel<<<4096, 256, 0, stream>>>(y2, (long)BATCH * 64 * 96, 96, counts2);
  bn_sign_params_kernel<<<1, 256, 0, stream>>>(counts2, bn2g, bn2b, sc2, sh2,
                                               96, 4096.0f * 64.0f);
  feat_stats_kernel<<<1536, 256, 0, stream>>>(y2, sc2, sh2, sum3, ssq3);
  bn3_params_kernel<<<6, 256, 0, stream>>>(sum3, ssq3, bn3g, bn3b, sc3, sh3);
  bn3_apply_kernel<<<24576, 256, 0, stream>>>(y2, sc2, sh2, sc3, sh3, h3);
  fc1_wmma_kernel<<<512, 256, 0, stream>>>(h3, w1h, f1b, y3);
  count_pos_kernel<<<2048, 256, 0, stream>>>(y3, (long)BATCH * 1024, 1024, counts4);
  bn_sign_params_kernel<<<4, 256, 0, stream>>>(counts4, bn4g, bn4b, sc4, sh4,
                                               1024, 4096.0f);
  fc2_kernel<<<160, 256, 0, stream>>>(y3, sc4, sh4, f2w, f2b, out);

  (void)in_sizes; (void)n_in; (void)out_size; (void)ws_size;
}
